// Voxel_MAE_49512382988387
// MI455X (gfx1250) — compile-verified, matrix operand order checked by reading, not run on hardware
//
#include <hip/hip_runtime.h>

typedef __attribute__((ext_vector_type(16))) _Float16 v16h;
typedef __attribute__((ext_vector_type(8)))  _Float16 v8h;
typedef __attribute__((ext_vector_type(8)))  float    v8f;
typedef __attribute__((ext_vector_type(4)))  float    v4f;

#define BATCH 2
#define BN_EPS 1e-3f

// Hardware K-slot mapping for 16-bit A/B fragments (CDNA5 WMMA 16x16x32):
// lane group g = lane/16, half h: h<8 -> k = g*8 + h ; h>=8 -> k = 16 + g*8 + (h-8)
__device__ __host__ __forceinline__ int hwK(int g, int h) {
  int v = h >> 1, w = h & 1;
  return ((v >= 4) ? 16 : 0) + g * 8 + ((v & 3) << 1) + w;
}

// ---------------- scatter voxels into dense f16 grid + uint8 mask ----------------
__global__ void k_scatter(const float* __restrict__ vf, const int* __restrict__ co,
                          _Float16* __restrict__ x0, unsigned char* __restrict__ m0, int n) {
  int i = blockIdx.x * blockDim.x + threadIdx.x;
  if (i >= n) return;
  int b = co[4 * i + 0], z = co[4 * i + 1], y = co[4 * i + 2], x = co[4 * i + 3];
  size_t idx = (((size_t)b * 32 + z) * 256 + y) * 256 + x;
  x0[idx] = (_Float16)vf[i];
  m0[idx] = 1;
}

// ---------------- stride-2 mask downsample: OR over 3x3x3 neighborhood ----------------
__global__ void k_mask_down(const unsigned char* __restrict__ mi, unsigned char* __restrict__ mo,
                            int Di, int Hi, int Wi, int Do, int Ho, int Wo) {
  int i = blockIdx.x * blockDim.x + threadIdx.x;
  int tot = BATCH * Do * Ho * Wo;
  if (i >= tot) return;
  int x = i % Wo; int t = i / Wo;
  int y = t % Ho; t /= Ho;
  int z = t % Do; int b = t / Do;
  unsigned char any = 0;
  for (int kz = 0; kz < 3; ++kz) {
    int zi = 2 * z - 1 + kz; if (zi < 0 || zi >= Di) continue;
    for (int ky = 0; ky < 3; ++ky) {
      int yi = 2 * y - 1 + ky; if (yi < 0 || yi >= Hi) continue;
      for (int kx = 0; kx < 3; ++kx) {
        int xi = 2 * x - 1 + kx; if (xi < 0 || xi >= Wi) continue;
        any |= mi[(((size_t)b * Di + zi) * Hi + yi) * Wi + xi];
      }
    }
  }
  mo[i] = any ? 1 : 0;
}

// ---- fold BN scale into f16 weights, laid out in per-lane fragment (hwK) order ----
__global__ void k_pack(const float* __restrict__ w, const float* __restrict__ ga,
                       const float* __restrict__ va, _Float16* __restrict__ wpk,
                       int Cin, int Cout, int ks, int Kpad) {
  int idx = blockIdx.x * blockDim.x + threadIdx.x;
  int nChunks = Kpad >> 5;
  int total = Cout * Kpad;
  if (idx >= total) return;
  int h = idx & 15;
  int lane = (idx >> 4) & 31;
  int cq = idx >> 9;
  int q = cq % nChunks;
  int nt = cq / nChunks;
  int g = lane >> 4, nn = lane & 15;
  int k = q * 32 + hwK(g, h);
  int n = nt * 16 + nn;
  int K = Cin * ks * ks * ks;
  float val = 0.f;
  if (k < K) {
    float s = ga[n] * rsqrtf(va[n] + BN_EPS);
    val = w[(size_t)n * K + k] * s;
  }
  wpk[idx] = (_Float16)val;
}

__global__ void k_bias(const float* __restrict__ ga, const float* __restrict__ be,
                       const float* __restrict__ me, const float* __restrict__ va,
                       float* __restrict__ bias, int C) {
  int i = threadIdx.x;
  if (i < C) {
    float s = ga[i] * rsqrtf(va[i] + BN_EPS);
    bias[i] = be[i] - me[i] * s;
  }
}

// ---------------- implicit-GEMM conv3d + fused BN bias + ReLU + mask ----------------
// Block: 16 consecutive x outputs at fixed (b,z,y); one wave per 16-out-channel tile.
// Stage: im2col receptive field into LDS A-tile [16][Kstride] (K contiguous); the
// K loop is then pure: 2x ds_load_b128 (A) + 1x 32B global load (B) + v_wmma.
template <int CIN, int KS, int ST, bool OUTF32>
__global__ void k_conv(const _Float16* __restrict__ inp, void* __restrict__ outp,
                       const _Float16* __restrict__ wpk, const float* __restrict__ bias,
                       const unsigned char* __restrict__ mk,
                       int Cout, int Di, int Hi, int Wi, int Do, int Ho, int Wo) {
  constexpr int K = CIN * KS * KS * KS;
  constexpr int Kpad = (K + 31) & ~31;
  constexpr int Kstride = Kpad + 8;             // +16B row pad to break LDS bank conflicts
  constexpr int PAD = (KS == 3) ? 1 : 0;
  constexpr int PW = 16 * ST + ((KS == 3) ? 2 : 0);
  constexpr int ROWS = CIN * KS * KS;
  constexpr int nChunks = Kpad / 32;

  __shared__ __attribute__((aligned(16))) _Float16 Atile[16 * Kstride];

  const int xt = blockIdx.x;
  const int y0 = blockIdx.y;
  const int z0 = blockIdx.z % Do;
  const int b  = blockIdx.z / Do;

  // ---- im2col stage: one (c,kz,ky) input row per iteration ----
  for (int r = threadIdx.x; r < ROWS; r += blockDim.x) {
    int c, kz, ky;
    if (KS == 3) { c = r / 9; int t = r - c * 9; kz = t / 3; ky = t - kz * 3; }
    else         { c = r; kz = 0; ky = 0; }
    int zi = z0 * ST - PAD + kz;
    int yi = y0 * ST - PAD + ky;
    bool zy = (zi >= 0 && zi < Di && yi >= 0 && yi < Hi);
    const _Float16* src =
        inp + (((size_t)(b * CIN + c) * Di + (zy ? zi : 0)) * Hi + (zy ? yi : 0)) * Wi;
    int xbase = xt * 16 * ST - PAD;
    for (int xx = 0; xx < PW; ++xx) {
      int xi = xbase + xx;
      _Float16 v = (_Float16)0.f;
      if (zy && xi >= 0 && xi < Wi) v = src[xi];
#pragma unroll
      for (int kx = 0; kx < KS; ++kx) {        // scatter value to (m,kx) users
        int t2 = xx - kx;
        if (t2 >= 0 && (t2 & (ST - 1)) == 0) {
          int m = t2 >> (ST / 2);              // ST=1 -> >>0, ST=2 -> >>1
          if (m < 16) Atile[m * Kstride + r * KS + kx] = v;
        }
      }
    }
  }
  if constexpr (K < Kpad) {                    // zero-fill K..Kpad tail
    constexpr int padc = Kpad - K;
    for (int i = threadIdx.x; i < 16 * padc; i += blockDim.x) {
      int m = i / padc;
      int k = K + (i - m * padc);
      Atile[m * Kstride + k] = (_Float16)0.f;
    }
  }
  __syncthreads();

  const int wave = threadIdx.x >> 5;           // N-tile index
  const int lane = threadIdx.x & 31;
  const int g  = lane >> 4;
  const int mn = lane & 15;                    // A: row M;  B/D: column N

  const _Float16* Abase = &Atile[mn * Kstride + g * 8];
  const _Float16* Bbase = wpk + (size_t)wave * nChunks * 512 + lane * 16;

  v8f acc = {0.f, 0.f, 0.f, 0.f, 0.f, 0.f, 0.f, 0.f};
#pragma unroll
  for (int q = 0; q < nChunks; ++q) {
    v8h lo = *(const v8h*)(Abase + q * 32);        // k = q*32 + g*8 + {0..7}
    v8h hi = *(const v8h*)(Abase + q * 32 + 16);   // k = q*32 + 16 + g*8 + {0..7}
    v16h afrag = __builtin_shufflevector(lo, hi, 0, 1, 2, 3, 4, 5, 6, 7,
                                         8, 9, 10, 11, 12, 13, 14, 15);
    v16h bfrag = *(const v16h*)(Bbase + (size_t)q * 512);
    acc = __builtin_amdgcn_wmma_f32_16x16x32_f16(false, afrag, false, bfrag,
                                                 (short)0, acc, false, false);
  }

  // ---- epilogue: lane owns 8 consecutive x outputs (M = 8g..8g+7) ----
  const int n = wave * 16 + mn;
  const float bn = bias[n];
  const size_t ob = (((size_t)(b * Cout + n) * Do + z0) * Ho + y0) * Wo + xt * 16 + 8 * g;
  const size_t mb = (((size_t)b * Do + z0) * Ho + y0) * Wo + xt * 16 + 8 * g;

  unsigned long long msk = *(const unsigned long long*)(mk + mb);  // 8 mask bytes, 8B-aligned
  float vals[8];
#pragma unroll
  for (int r = 0; r < 8; ++r) {
    float vv = fmaxf(acc[r] + bn, 0.f);
    vals[r] = ((msk >> (8 * r)) & 0xffULL) ? vv : 0.f;             // branchless cndmask
  }
  if constexpr (OUTF32) {
    v4f lo = {vals[0], vals[1], vals[2], vals[3]};
    v4f hi = {vals[4], vals[5], vals[6], vals[7]};
    *(v4f*)((float*)outp + ob)     = lo;                           // 2x global_store_b128
    *(v4f*)((float*)outp + ob + 4) = hi;
  } else {
    v8h o;
#pragma unroll
    for (int r = 0; r < 8; ++r) o[r] = (_Float16)vals[r];
    *(v8h*)((_Float16*)outp + ob) = o;                             // 1x global_store_b128
  }
}

// ------------------------------------ host ------------------------------------
extern "C" void kernel_launch(void* const* d_in, const int* in_sizes, int n_in,
                              void* d_out, int out_size, void* d_ws, size_t ws_size,
                              hipStream_t stream) {
  (void)out_size; (void)ws_size;
  if (n_in < 62) return;
  const float* vf = (const float*)d_in[0];
  const int* coords = (const int*)d_in[1];
  const int nvox = in_sizes[0];

  // ---- resolve params leaf order (sorted-keys vs insertion-order pytree) ----
  static const int mapS[12] = {10, 0, 1, 2, 3, 4, 5, 6, 7, 8, 9, 11};
  static const int mapI[12] = {0, 1, 2, 3, 4, 5, 6, 7, 8, 9, 10, 11};
  int b0 = 2;
  bool bf0 = (in_sizes[b0] == in_sizes[b0 + 1]);
  int w0 = bf0 ? in_sizes[b0 + 4] : in_sizes[b0];
  const int* map = (w0 == 432) ? mapI : mapS;  // conv_input.w = 16*1*27 = 432

  const float *Wt[12], *Ga[12], *Be[12], *Me[12], *Va[12];
  for (int l = 0; l < 12; ++l) {
    int gi = map[l];
    int base = 2 + 5 * gi;
    bool bf = (in_sizes[base] == in_sizes[base + 1]);
    if (bf) {
      Ga[l] = (const float*)d_in[base + 0]; Be[l] = (const float*)d_in[base + 1];
      Me[l] = (const float*)d_in[base + 2]; Va[l] = (const float*)d_in[base + 3];
      Wt[l] = (const float*)d_in[base + 4];
    } else {
      Wt[l] = (const float*)d_in[base + 0];
      Ga[l] = (const float*)d_in[base + 1]; Be[l] = (const float*)d_in[base + 2];
      Me[l] = (const float*)d_in[base + 3]; Va[l] = (const float*)d_in[base + 4];
    }
  }

  // ---- network topology ----
  static const int Cin[12]  = {1, 16, 16, 32, 32, 32, 64, 64, 64, 64, 64, 64};
  static const int Cout[12] = {16, 16, 32, 32, 32, 64, 64, 64, 64, 64, 64, 128};
  static const int KS[12]   = {3, 3, 3, 3, 3, 3, 3, 3, 3, 3, 3, 1};
  static const int LVI[12]  = {0, 0, 0, 1, 1, 1, 2, 2, 2, 3, 3, 3};
  static const int LVO[12]  = {0, 0, 1, 1, 1, 2, 2, 2, 3, 3, 3, 3};
  static const int DIM[4][3] = {{32, 256, 256}, {16, 128, 128}, {8, 64, 64}, {4, 32, 32}};

  // ---- workspace: two ping-pong activation regions + masks + packed weights ----
  char* ws = (char*)d_ws;
  size_t off = 0;
  auto bump = [&](size_t bytes) -> char* {
    char* p = ws + off;
    off = (off + bytes + 255) & ~(size_t)255;
    return p;
  };
  const size_t RBYTES = (size_t)BATCH * 16 * 32 * 256 * 256 * 2;  // 134 MB
  char* R0 = bump(RBYTES);
  char* R1 = bump(RBYTES);
  unsigned char* mask[4];
  for (int lv = 0; lv < 4; ++lv)
    mask[lv] = (unsigned char*)bump((size_t)BATCH * DIM[lv][0] * DIM[lv][1] * DIM[lv][2]);
  _Float16* wp[12]; float* bias[12]; int Kpad[12];
  for (int l = 0; l < 12; ++l) {
    int K = Cin[l] * KS[l] * KS[l] * KS[l];
    Kpad[l] = (K + 31) & ~31;
    wp[l] = (_Float16*)bump((size_t)Cout[l] * Kpad[l] * 2);
    bias[l] = (float*)bump((size_t)Cout[l] * 4);
  }

  // ---- launch sequence (all on stream; graph-capture safe) ----
  hipMemsetAsync(R0, 0, (size_t)BATCH * 1 * 32 * 256 * 256 * 2, stream);
  hipMemsetAsync(mask[0], 0, (size_t)BATCH * 32 * 256 * 256, stream);
  k_scatter<<<(nvox + 255) / 256, 256, 0, stream>>>(vf, coords, (_Float16*)R0, mask[0], nvox);

  for (int lv = 0; lv < 3; ++lv) {
    int Do = DIM[lv + 1][0], Ho = DIM[lv + 1][1], Wo = DIM[lv + 1][2];
    int tot = BATCH * Do * Ho * Wo;
    k_mask_down<<<(tot + 255) / 256, 256, 0, stream>>>(
        mask[lv], mask[lv + 1], DIM[lv][0], DIM[lv][1], DIM[lv][2], Do, Ho, Wo);
  }

  for (int l = 0; l < 12; ++l) {
    k_bias<<<1, Cout[l], 0, stream>>>(Ga[l], Be[l], Me[l], Va[l], bias[l], Cout[l]);
    int tot = Cout[l] * Kpad[l];
    k_pack<<<(tot + 255) / 256, 256, 0, stream>>>(Wt[l], Ga[l], Va[l], wp[l],
                                                  Cin[l], Cout[l], KS[l], Kpad[l]);
  }

  for (int l = 0; l < 12; ++l) {
    const _Float16* inp = (const _Float16*)((l % 2 == 0) ? R0 : R1);
    void* outp = (l == 11) ? d_out : (void*)((l % 2 == 0) ? R1 : R0);
    int Di = DIM[LVI[l]][0], Hi = DIM[LVI[l]][1], Wi = DIM[LVI[l]][2];
    int Do = DIM[LVO[l]][0], Ho = DIM[LVO[l]][1], Wo = DIM[LVO[l]][2];
    dim3 grid(Wo / 16, Ho, Do * BATCH);
    dim3 blk(32 * (Cout[l] / 16));
    const unsigned char* mkp = mask[LVO[l]];
#define LAUNCH(CI, KK, SS, F32)                                                       \
  k_conv<CI, KK, SS, F32><<<grid, blk, 0, stream>>>(inp, outp, wp[l], bias[l], mkp,   \
                                                    Cout[l], Di, Hi, Wi, Do, Ho, Wo)
    switch (l) {
      case 0:  LAUNCH(1, 3, 1, false);  break;
      case 1:  LAUNCH(16, 3, 1, false); break;
      case 2:  LAUNCH(16, 3, 2, false); break;
      case 3:  LAUNCH(32, 3, 1, false); break;
      case 4:  LAUNCH(32, 3, 1, false); break;
      case 5:  LAUNCH(32, 3, 2, false); break;
      case 6:  LAUNCH(64, 3, 1, false); break;
      case 7:  LAUNCH(64, 3, 1, false); break;
      case 8:  LAUNCH(64, 3, 2, false); break;
      case 9:  LAUNCH(64, 3, 1, false); break;
      case 10: LAUNCH(64, 3, 1, false); break;
      case 11: LAUNCH(64, 1, 1, true);  break;
    }
#undef LAUNCH
  }
}
